// GCNN_6880537608256
// MI455X (gfx1250) — matrix operand order repaired and checked by reading, not compiled, not run on hardware
//
#include <hip/hip_runtime.h>
#include <hip/hip_bf16.h>

// ---------------------------------------------------------------------------
// CDNA5 WMMA (wave32) types: bf16 A/B fragments (8 VGPRs, 16 packed elems),
// f32 C/D accumulators (8 VGPRs).
// ---------------------------------------------------------------------------
typedef __attribute__((ext_vector_type(16))) __bf16 v16bf;
typedef __attribute__((ext_vector_type(8)))  float  v8f;

// ---------------------------------------------------------------------------
// A fragment (16x32 bf16, ISA layout) from fp32 row-major A.
//   lanes 0-15 : M = lane,    K runs {kb..kb+7} and {kb+16..kb+23}, kb = k0
//   lanes 16-31: M = lane-16, same with kb = k0+8
// Two contiguous 8-float runs per lane -> 4x float4 loads + 8x v_cvt_pk.
// p = &A[m*K], kb already includes the lane-half offset.
// ---------------------------------------------------------------------------
__device__ __forceinline__ v16bf frag_a(const float* __restrict__ p, int kb) {
    const float4 f0 = *(const float4*)(p + kb + 0);
    const float4 f1 = *(const float4*)(p + kb + 4);
    const float4 f2 = *(const float4*)(p + kb + 16);
    const float4 f3 = *(const float4*)(p + kb + 20);
    v16bf a;
    a[0]  = (__bf16)f0.x; a[1]  = (__bf16)f0.y; a[2]  = (__bf16)f0.z; a[3]  = (__bf16)f0.w;
    a[4]  = (__bf16)f1.x; a[5]  = (__bf16)f1.y; a[6]  = (__bf16)f1.z; a[7]  = (__bf16)f1.w;
    a[8]  = (__bf16)f2.x; a[9]  = (__bf16)f2.y; a[10] = (__bf16)f2.z; a[11] = (__bf16)f2.w;
    a[12] = (__bf16)f3.x; a[13] = (__bf16)f3.y; a[14] = (__bf16)f3.z; a[15] = (__bf16)f3.w;
    return a;
}

// ---------------------------------------------------------------------------
// Pack fp32 weights W [Hout x K] (row-major, used as out = A @ W^T) into bf16
// B fragments in WMMA lane order. One thread = one (n-tile, k-tile, lane)
// fragment = 16 contiguous bf16 (32B).
// B layout: lane l -> N = nt*16 + (l&15); K = kt*32 + ((l&16)?8:0) + i + ((i&8)?8:0).
// ---------------------------------------------------------------------------
__global__ __launch_bounds__(256)
void gcnn_pack_w(const float* __restrict__ W, __bf16* __restrict__ Wp,
                 int Hout, int K) {
    const int t = blockIdx.x * blockDim.x + threadIdx.x;
    const int total = (Hout >> 4) * (K >> 5) * 32;
    if (t >= total) return;
    const int lane = t & 31;
    const int kt   = (t >> 5) % (K >> 5);
    const int nt   = (t >> 5) / (K >> 5);
    const int n    = nt * 16 + (lane & 15);
    const int kb   = kt * 32 + ((lane & 16) ? 8 : 0);
    const float* p = W + (long long)n * K + kb;
    v16bf q;
#pragma unroll
    for (int i = 0; i < 16; ++i) q[i] = (__bf16)p[i + ((i & 8) ? 8 : 0)];
    *(v16bf*)(Wp + (long long)t * 16) = q;
}

// ---------------------------------------------------------------------------
// Fused GraphConv GEMM: out = act( A1 @ W1^T + A2 @ W2^T + bias )
// A1/A2 fp32 [Nrows x K]; W1p/W2p pre-packed bf16 fragments.
// Block = 256 threads (8 waves). Each wave owns a 16-row strip; the block
// covers 64 output cols (4 N-tiles) whose packed B fragments are staged in
// LDS once and reused by all 8 waves. 4 WMMAs per K-step per weight matrix.
// Dynamic LDS = 2 * 4 * (K/32)*512 bf16 = K*256 bytes.
// ---------------------------------------------------------------------------
__global__ __launch_bounds__(256)
void gcnn_gemm_wmma(const float* __restrict__ A1, const __bf16* __restrict__ W1p,
                    const float* __restrict__ A2, const __bf16* __restrict__ W2p,
                    const float* __restrict__ bias, float* __restrict__ out,
                    int Nrows, int K, int Hout, int do_relu) {
    extern __shared__ __bf16 ldsb[];
    const int ktiles    = K >> 5;            // K / 32
    const int tileElems = ktiles * 512;      // bf16 per 16-col B tile
    const int lane = threadIdx.x & 31;
    const int wave = threadIdx.x >> 5;
    const int nt0  = blockIdx.y * 4;         // first 16-col tile of this block

    // ---- stage packed B tiles (both weight matrices) into LDS ----
    {
        const int vecs = (4 * tileElems) >> 3;   // int4 (= 8 bf16) units each
        const int4* g1 = (const int4*)(W1p + (long long)nt0 * tileElems);
        const int4* g2 = (const int4*)(W2p + (long long)nt0 * tileElems);
        int4* l1 = (int4*)ldsb;
        int4* l2 = (int4*)(ldsb + 4 * tileElems);
        for (int t = threadIdx.x; t < vecs; t += 256) { l1[t] = g1[t]; l2[t] = g2[t]; }
    }
    __syncthreads();                          // all waves participate

    const int tileM = (blockIdx.x * 8 + wave) * 16;   // wave-uniform guard
    if (tileM < Nrows) {
        v8f acc[4] = {};
        const int m     = tileM + (lane & 15);
        const int kbOff = (lane & 16) ? 8 : 0;         // A lane-half K offset
        const float* pa1 = A1 + (long long)m * K;
        const float* pa2 = A2 + (long long)m * K;
        const v16bf* lb1 = (const v16bf*)ldsb;                      // [j*ktiles*32 + kt*32 + lane]
        const v16bf* lb2 = (const v16bf*)(ldsb + 4 * tileElems);

        for (int kt = 0; kt < ktiles; ++kt) {
            __builtin_prefetch(pa1 + kt * 32 + 256, 0, 0);          // global_prefetch_b8
            const v16bf a = frag_a(pa1, kt * 32 + kbOff);
            const int bi = kt * 32 + lane;
#pragma unroll
            for (int j = 0; j < 4; ++j) {
                const v16bf b = lb1[j * ktiles * 32 + bi];
                acc[j] = __builtin_amdgcn_wmma_f32_16x16x32_bf16(
                             false, a, false, b, (short)0, acc[j], false, false);
            }
        }
        for (int kt = 0; kt < ktiles; ++kt) {
            __builtin_prefetch(pa2 + kt * 32 + 256, 0, 0);
            const v16bf a = frag_a(pa2, kt * 32 + kbOff);
            const int bi = kt * 32 + lane;
#pragma unroll
            for (int j = 0; j < 4; ++j) {
                const v16bf b = lb2[j * ktiles * 32 + bi];
                acc[j] = __builtin_amdgcn_wmma_f32_16x16x32_bf16(
                             false, a, false, b, (short)0, acc[j], false, false);
            }
        }

        // C/D layout: VGPR r -> M = r (lanes 0-15) / 8+r (lanes 16-31); N = lane&15
        const int mb = tileM + ((lane & 16) ? 8 : 0);
#pragma unroll
        for (int j = 0; j < 4; ++j) {
            const int n  = (nt0 + j) * 16 + (lane & 15);
            const float bv = bias ? bias[n] : 0.0f;
#pragma unroll
            for (int r = 0; r < 8; ++r) {
                float v = acc[j][r] + bv;
                if (do_relu) v = fmaxf(v, 0.0f);
                out[(long long)(mb + r) * Hout + n] = v;
            }
        }
    }
}

// ---------------------------------------------------------------------------
// Edge scatter: aggr[dst] += w_e * x[src]. Consecutive lanes cover one edge's
// feature row (float4 gather + 4 contiguous f32 atomics). h1/aggr buffers are
// L2-resident (102.4 MB < 192 MB L2) so the random traffic stays on-die.
// ---------------------------------------------------------------------------
__global__ __launch_bounds__(256)
void gcnn_edge_scatter(const float* __restrict__ x,
                       const long long* __restrict__ src,
                       const long long* __restrict__ dst,
                       const float* __restrict__ w,
                       float* __restrict__ aggr, int E, int D) {
    const int D4 = D >> 2;
    const long long total  = (long long)E * D4;
    const long long stride = (long long)gridDim.x * blockDim.x;
    for (long long i = (long long)blockIdx.x * blockDim.x + threadIdx.x;
         i < total; i += stride) {
        const int e  = (int)(i / D4);
        const int f4 = (int)(i - (long long)e * D4);
        const int s  = (int)src[e];
        const int d  = (int)dst[e];
        const float we = w[e];
        const float4 xv = ((const float4*)(x + (long long)s * D))[f4];
        float* ap = aggr + (long long)d * D + 4 * f4;
        atomicAdd(ap + 0, we * xv.x);
        atomicAdd(ap + 1, we * xv.y);
        atomicAdd(ap + 2, we * xv.z);
        atomicAdd(ap + 3, we * xv.w);
    }
}

// ---------------------------------------------------------------------------
// Global mean-pool stage 1: per-graph sums + counts via atomics (G=64).
// ---------------------------------------------------------------------------
__global__ __launch_bounds__(256)
void gcnn_pool_sum(const float* __restrict__ h,
                   const long long* __restrict__ batch,
                   float* __restrict__ sums, float* __restrict__ cnts,
                   int N, int H) {
    const long long total  = (long long)N * H;
    const long long stride = (long long)gridDim.x * blockDim.x;
    for (long long i = (long long)blockIdx.x * blockDim.x + threadIdx.x;
         i < total; i += stride) {
        const int n = (int)(i / H);
        const int f = (int)(i - (long long)n * H);
        const int g = (int)batch[n];
        atomicAdd(&sums[(long long)g * H + f], h[i]);
        if (f == 0) atomicAdd(&cnts[g], 1.0f);
    }
}

// ---------------------------------------------------------------------------
// MLP head: one block per graph, 512->64->16->1 (~2M MACs total, negligible).
// ---------------------------------------------------------------------------
__global__ __launch_bounds__(256)
void gcnn_mlp_head(const float* __restrict__ pooled, const float* __restrict__ cnts,
                   const float* __restrict__ Wl1, const float* __restrict__ bl1,
                   const float* __restrict__ Wl2, const float* __restrict__ bl2,
                   const float* __restrict__ Wl3, const float* __restrict__ bl3,
                   float* __restrict__ out, int H) {
    __shared__ float p[512];
    __shared__ float t1[64];
    __shared__ float t2[16];
    const int g = blockIdx.x;
    const float inv = 1.0f / fmaxf(cnts[g], 1.0f);
    for (int i = threadIdx.x; i < H; i += blockDim.x)
        p[i] = pooled[(long long)g * H + i] * inv;
    __syncthreads();
    for (int o = threadIdx.x; o < 64; o += blockDim.x) {
        float s = bl1[o];
        const float* wr = Wl1 + (long long)o * H;
        for (int k = 0; k < H; ++k) s += p[k] * wr[k];
        t1[o] = fmaxf(s, 0.0f);
    }
    __syncthreads();
    for (int o = threadIdx.x; o < 16; o += blockDim.x) {
        float s = bl2[o];
        const float* wr = Wl2 + o * 64;
        for (int k = 0; k < 64; ++k) s += t1[k] * wr[k];
        t2[o] = fmaxf(s, 0.0f);
    }
    __syncthreads();
    if (threadIdx.x == 0) {
        float s = bl3[0];
        for (int k = 0; k < 16; ++k) s += t2[k] * Wl3[k];
        out[g] = s;
    }
}

// ---------------------------------------------------------------------------
// Host-side orchestration.
// Inputs: x, edge_index, edge_attr, batch, W1_rel, b1_rel, W1_root,
//         W2_rel, b2_rel, W2_root, Wl1, bl1, Wl2, bl2, Wl3, bl3
// ---------------------------------------------------------------------------
extern "C" void kernel_launch(void* const* d_in, const int* in_sizes, int n_in,
                              void* d_out, int out_size, void* d_ws, size_t ws_size,
                              hipStream_t stream) {
    const float*     x       = (const float*)d_in[0];
    const long long* eidx    = (const long long*)d_in[1];   // [2,E] int64
    const float*     eattr   = (const float*)d_in[2];
    const long long* batch   = (const long long*)d_in[3];
    const float*     W1_rel  = (const float*)d_in[4];
    const float*     b1_rel  = (const float*)d_in[5];
    const float*     W1_root = (const float*)d_in[6];
    const float*     W2_rel  = (const float*)d_in[7];
    const float*     b2_rel  = (const float*)d_in[8];
    const float*     W2_root = (const float*)d_in[9];
    const float*     Wl1     = (const float*)d_in[10];
    const float*     bl1     = (const float*)d_in[11];
    const float*     Wl2     = (const float*)d_in[12];
    const float*     bl2     = (const float*)d_in[13];
    const float*     Wl3     = (const float*)d_in[14];
    const float*     bl3     = (const float*)d_in[15];
    float* out = (float*)d_out;

    const int N = in_sizes[3];               // batch is [N]      -> 50000
    const int E = in_sizes[2];               // edge_attr is [E]  -> 800000
    const int F = in_sizes[0] / N;           // 128
    const int H = in_sizes[5];               // b1_rel -> 512
    const int G = out_size;                  // 64

    const long long* src = eidx;
    const long long* dst = eidx + E;

    // Workspace layout (256B aligned slices)
    size_t off = 0;
    auto carve = [&](size_t bytes) {
        void* p = (char*)d_ws + off;
        off += (bytes + 255) & ~(size_t)255;
        return p;
    };
    float*  aggr1   = (float*)carve((size_t)N * F * sizeof(float));
    float*  h1      = (float*)carve((size_t)N * H * sizeof(float));
    float*  aggr2   = (float*)carve((size_t)N * H * sizeof(float));
    float*  h2      = (float*)carve((size_t)N * H * sizeof(float));
    float*  pooled  = (float*)carve((size_t)G * H * sizeof(float));
    float*  cnts    = (float*)carve((size_t)G * sizeof(float));
    __bf16* W1relP  = (__bf16*)carve((size_t)H * F * sizeof(__bf16));
    __bf16* W1rootP = (__bf16*)carve((size_t)H * F * sizeof(__bf16));
    __bf16* W2relP  = (__bf16*)carve((size_t)H * H * sizeof(__bf16));
    __bf16* W2rootP = (__bf16*)carve((size_t)H * H * sizeof(__bf16));
    (void)ws_size;

    // Zero accumulation buffers every call (graph-capture-safe async memsets)
    hipMemsetAsync(aggr1,  0, (size_t)N * F * sizeof(float), stream);
    hipMemsetAsync(aggr2,  0, (size_t)N * H * sizeof(float), stream);
    hipMemsetAsync(pooled, 0, (size_t)G * H * sizeof(float), stream);
    hipMemsetAsync(cnts,   0, (size_t)G * sizeof(float), stream);

    // ---- pre-pack weights into WMMA-fragment bf16 layout (tiny, L2-hot) ----
    {
        const int t1n = (H / 16) * (F / 32) * 32;       // 4096 threads
        const int t2n = (H / 16) * (H / 32) * 32;       // 16384 threads
        gcnn_pack_w<<<(t1n + 255) / 256, 256, 0, stream>>>(W1_rel,  W1relP,  H, F);
        gcnn_pack_w<<<(t1n + 255) / 256, 256, 0, stream>>>(W1_root, W1rootP, H, F);
        gcnn_pack_w<<<(t2n + 255) / 256, 256, 0, stream>>>(W2_rel,  W2relP,  H, H);
        gcnn_pack_w<<<(t2n + 255) / 256, 256, 0, stream>>>(W2_root, W2rootP, H, H);
    }

    const int rowTiles = (N + 15) / 16;                 // 3125
    dim3 gemmBlk(256, 1, 1);

    // ---- Layer 1 ----
    gcnn_edge_scatter<<<8192, 256, 0, stream>>>(x, src, dst, eattr, aggr1, E, F);
    {
        dim3 grid((rowTiles + 7) / 8, H / 64, 1);
        const size_t shmem = (size_t)F * 256;           // 2 * 4 * (F/32)*512 * 2B
        gcnn_gemm_wmma<<<grid, gemmBlk, shmem, stream>>>(aggr1, W1relP, x, W1rootP,
                                                         b1_rel, h1, N, F, H, 1);
    }

    // ---- Layer 2 ----
    gcnn_edge_scatter<<<16384, 256, 0, stream>>>(h1, src, dst, eattr, aggr2, E, H);
    {
        dim3 grid((rowTiles + 7) / 8, H / 64, 1);
        const size_t shmem = (size_t)H * 256;           // 128 KB (<= 320 KB WGP LDS)
        gcnn_gemm_wmma<<<grid, gemmBlk, shmem, stream>>>(aggr2, W2relP, h1, W2rootP,
                                                         b2_rel, h2, N, H, H, 1);
    }

    // ---- Pool + head ----
    gcnn_pool_sum<<<8192, 256, 0, stream>>>(h2, batch, pooled, cnts, N, H);
    gcnn_mlp_head<<<G, 256, 0, stream>>>(pooled, cnts, Wl1, bl1, Wl2, bl2,
                                         Wl3, bl3, out, H);
}
